// WarCraftModel_31104153157789
// MI455X (gfx1250) — matrix operand order repaired and checked by reading, not compiled, not run on hardware
//
#include <hip/hip_runtime.h>
#include <hip/hip_bf16.h>

// ---------------------------------------------------------------------------
// 2-layer GCN forward for MI455X (gfx1250, wave32, WMMA).
// Dense transforms: v_wmma_f32_16x16x32_f16, one wave per 16-node tile.
// Edge aggregation: per-edge norm precomputed; wave-per-edge scatter with
// scalar (readfirstlane) index loads, coalesced 128B row gather, and 32
// consecutive global_atomic_add_f32 per wave.
// ---------------------------------------------------------------------------

typedef __attribute__((ext_vector_type(16))) _Float16 v16h;
typedef __attribute__((ext_vector_type(8)))  float    v8f;

#define BN_EPS 1e-5f

// ---------------------------- degree / norm --------------------------------

__global__ void k_deg_init(float* deg, int n) {
    int i = blockIdx.x * blockDim.x + threadIdx.x;
    if (i < n) deg[i] = 1.0f;                 // self-loop weight
}

__global__ void k_deg_accum(const int* __restrict__ col,
                            const float* __restrict__ ew,
                            float* deg, int e) {
    int i = blockIdx.x * blockDim.x + threadIdx.x;
    if (i < e) atomicAdd(&deg[col[i]], ew[i]);
}

__global__ void k_dinv(float* deg, int n) {   // in place: deg -> dinv
    int i = blockIdx.x * blockDim.x + threadIdx.x;
    if (i < n) {
        float d = deg[i];
        deg[i] = (d > 0.0f) ? rsqrtf(d) : 0.0f;
    }
}

__global__ void k_norm(const int* __restrict__ row, const int* __restrict__ col,
                       const float* __restrict__ ew, const float* __restrict__ dinv,
                       float* __restrict__ nrm, int e) {
    int i = blockIdx.x * blockDim.x + threadIdx.x;
    if (i < e) nrm[i] = dinv[row[i]] * ew[i] * dinv[col[i]];
}

// ------------------------- layer 1: xw = x @ W1 (WMMA) ---------------------
// A: 16x32 f16 (rows = nodes, K = in-feature padded 3->32 with zeros).
// B: 32x16 f16 (W1 columns); two fragments for out cols 0-15 / 16-31.
// Fragment maps per CDNA5 ISA §7.12.2 (wave32).

__global__ void k_xw_wmma(const float* __restrict__ x,
                          const float* __restrict__ W1,
                          float* __restrict__ xw, int n, int ntiles) {
    int wave = (int)((blockIdx.x * blockDim.x + threadIdx.x) >> 5);
    int lane = threadIdx.x & 31;
    if (wave >= ntiles) return;               // wave-uniform exit (EXEC stays full)
    int base = wave * 16;
    int m  = lane & 15;
    int hi = lane >> 4;
    int node = base + m;

    v16h a;
#pragma unroll
    for (int i = 0; i < 16; ++i) {            // 16-bit A layout (16x32)
        int k = (i < 8) ? (i + 8 * hi) : (i + 8 + 8 * hi);
        float v = (k < 3 && node < n) ? x[node * 3 + k] : 0.0f;
        a[i] = (_Float16)v;
    }
    v16h b0, b1;
#pragma unroll
    for (int i = 0; i < 16; ++i) {            // B: K = (i<8 ? i : i+8) + 8*hi, N = m
        int k = (i < 8) ? (i + 8 * hi) : (i + 8 + 8 * hi);
        float w0 = (k < 3) ? W1[k * 32 + m]      : 0.0f;
        float w1 = (k < 3) ? W1[k * 32 + 16 + m] : 0.0f;
        b0[i] = (_Float16)w0;
        b1[i] = (_Float16)w1;
    }
    v8f c0 = {}, c1 = {};
    c0 = __builtin_amdgcn_wmma_f32_16x16x32_f16(false, a, false, b0, (short)0, c0, false, false);
    c1 = __builtin_amdgcn_wmma_f32_16x16x32_f16(false, a, false, b1, (short)0, c1, false, false);

    if (base + 16 <= n) {                     // full tile: wave-uniform fast path
#pragma unroll
        for (int r = 0; r < 8; ++r) {         // D: M = r + 8*hi, N = m
            int nd = base + r + 8 * hi;
            xw[nd * 32 + m]      = c0[r];
            xw[nd * 32 + 16 + m] = c1[r];
        }
    } else {                                  // tail tile: guarded stores
#pragma unroll
        for (int r = 0; r < 8; ++r) {
            int nd = base + r + 8 * hi;
            if (nd < n) {
                xw[nd * 32 + m]      = c0[r];
                xw[nd * 32 + 16 + m] = c1[r];
            }
        }
    }
}

// --------------------- aggregation 1 (self-loop + edges) -------------------

__global__ void k_agg1_init(const float* __restrict__ xw,
                            const float* __restrict__ dinv,
                            float* __restrict__ agg1, int n) {
    int i = blockIdx.x * blockDim.x + threadIdx.x;
    if (i < n * 32) {
        int nd = i >> 5;
        float di = dinv[nd];
        agg1[i] = di * di * xw[i];            // full overwrite (ws is poisoned)
    }
}

// wave-per-edge: readfirstlane -> scalar index/norm loads (s_load via K$);
// 32 lanes = 32 channels -> one coalesced 128B row read + 32 consecutive atomics.
__global__ void k_scatter1(const int* __restrict__ row, const int* __restrict__ col,
                           const float* __restrict__ nrm,
                           const float* __restrict__ xw,
                           float* __restrict__ agg1, int e) {
    int ed = __builtin_amdgcn_readfirstlane(
        (int)(((long long)blockIdx.x * blockDim.x + threadIdx.x) >> 5));
    int k = threadIdx.x & 31;
    if (ed < e) {                             // wave-uniform
        int r = row[ed], c = col[ed];         // scalar loads
        float w = nrm[ed];                    // scalar load
        atomicAdd(&agg1[c * 32 + k], w * xw[r * 32 + k]);
    }
}

// ------------------------------ batch norm ---------------------------------

__global__ void k_stats_zero(float* stats) {
    int i = threadIdx.x;
    if (i < 64) stats[i] = 0.0f;              // [0..31]=sum, [32..63]=sumsq
}

__global__ void k_bn_reduce(const float* __restrict__ agg1, float* stats, int n) {
    __shared__ float s_sum[32];
    __shared__ float s_sq[32];
    if (threadIdx.x < 32) { s_sum[threadIdx.x] = 0.0f; s_sq[threadIdx.x] = 0.0f; }
    __syncthreads();
    int k = threadIdx.x & 31;                 // stride multiple of 32 -> k fixed
    float ls = 0.0f, lq = 0.0f;
    long long total = (long long)n * 32;
    for (long long i = (long long)blockIdx.x * blockDim.x + threadIdx.x;
         i < total; i += (long long)gridDim.x * blockDim.x) {
        float v = agg1[i];
        ls += v; lq += v * v;
    }
    atomicAdd(&s_sum[k], ls);
    atomicAdd(&s_sq[k],  lq);
    __syncthreads();
    if (threadIdx.x < 32) {
        atomicAdd(&stats[threadIdx.x],      s_sum[threadIdx.x]);
        atomicAdd(&stats[32 + threadIdx.x], s_sq[threadIdx.x]);
    }
}

__global__ void k_bn_finalize(const float* __restrict__ stats,
                              const float* __restrict__ gamma,
                              const float* __restrict__ beta,
                              float* __restrict__ ss, int n) {
    int k = threadIdx.x;
    if (k < 32) {
        float invN = 1.0f / (float)n;
        float mean = stats[k] * invN;
        float var  = stats[32 + k] * invN - mean * mean;
        float sc   = gamma[k] * rsqrtf(var + BN_EPS);
        ss[k]      = sc;                      // scale
        ss[32 + k] = beta[k] - mean * sc;     // shift (b1 cancels through BN)
    }
}

// ------------- layer 2: hw = relu(bn(agg1)) @ W3 (fused, WMMA) -------------
// A loaded as 4x float4 (two contiguous 8-float K runs per lane), BN+ReLU in
// registers before f16 convert. B has only column N=0 nonzero (W3).

__global__ void k_hw_wmma(const float* __restrict__ agg1,
                          const float* __restrict__ ss,
                          const float* __restrict__ W3,
                          float* __restrict__ hw, int n, int ntiles) {
    int wave = (int)((blockIdx.x * blockDim.x + threadIdx.x) >> 5);
    int lane = threadIdx.x & 31;
    if (wave >= ntiles) return;
    int base = wave * 16;
    int m  = lane & 15;
    int hi = lane >> 4;
    int node = base + m;

    v16h a;
    if (node < n) {
        // lane covers K in [8hi, 8hi+8) and [16+8hi, 16+8hi+8)
        const float4* rowp = (const float4*)(agg1 + (size_t)node * 32);
        float4 q[4];
        q[0] = rowp[2 * hi];
        q[1] = rowp[2 * hi + 1];
        q[2] = rowp[4 + 2 * hi];
        q[3] = rowp[5 + 2 * hi];
        float v[16] = { q[0].x, q[0].y, q[0].z, q[0].w,
                        q[1].x, q[1].y, q[1].z, q[1].w,
                        q[2].x, q[2].y, q[2].z, q[2].w,
                        q[3].x, q[3].y, q[3].z, q[3].w };
#pragma unroll
        for (int i = 0; i < 16; ++i) {
            int k = (i < 8) ? (i + 8 * hi) : (i + 8 + 8 * hi);
            a[i] = (_Float16)fmaxf(0.0f, ss[k] * v[i] + ss[32 + k]);   // BN + ReLU
        }
    } else {
#pragma unroll
        for (int i = 0; i < 16; ++i) a[i] = (_Float16)0.0f;
    }

    v16h b;
#pragma unroll
    for (int i = 0; i < 16; ++i) {
        int k = (i < 8) ? (i + 8 * hi) : (i + 8 + 8 * hi);
        b[i] = (m == 0) ? (_Float16)W3[k] : (_Float16)0.0f;
    }
    v8f c = {};
    c = __builtin_amdgcn_wmma_f32_16x16x32_f16(false, a, false, b, (short)0, c, false, false);

    if (m == 0) {                             // column N=0 of D
        if (base + 16 <= n) {
#pragma unroll
            for (int r = 0; r < 8; ++r) hw[base + r + 8 * hi] = c[r];
        } else {
#pragma unroll
            for (int r = 0; r < 8; ++r) {
                int nd = base + r + 8 * hi;
                if (nd < n) hw[nd] = c[r];
            }
        }
    }
}

// --------------------- aggregation 2 (self-loop + edges) -------------------

__global__ void k_out_init(const float* __restrict__ hw,
                           const float* __restrict__ dinv,
                           const float* __restrict__ b3,
                           float* __restrict__ out, int n) {
    int i = blockIdx.x * blockDim.x + threadIdx.x;
    if (i < n) {
        float di = dinv[i];
        out[i] = di * di * hw[i] + b3[0];     // full overwrite of d_out
    }
}

__global__ void k_scatter2(const int* __restrict__ row, const int* __restrict__ col,
                           const float* __restrict__ nrm,
                           const float* __restrict__ hw,
                           float* __restrict__ out, int e) {
    int i = blockIdx.x * blockDim.x + threadIdx.x;
    if (i < e) atomicAdd(&out[col[i]], nrm[i] * hw[row[i]]);
}

// ------------------------------- launcher ----------------------------------

extern "C" void kernel_launch(void* const* d_in, const int* in_sizes, int n_in,
                              void* d_out, int out_size, void* d_ws, size_t ws_size,
                              hipStream_t stream) {
    (void)n_in; (void)out_size; (void)ws_size;

    const float* x      = (const float*)d_in[0];   // [n,3]
    const int*   ei     = (const int*)  d_in[1];   // [2,e]
    const float* ew     = (const float*)d_in[2];   // [e]
    const float* W1     = (const float*)d_in[3];   // [3,32]
    /* b1 = d_in[4] — cancels exactly through BatchNorm */
    const float* gamma1 = (const float*)d_in[5];   // [32]
    const float* beta1  = (const float*)d_in[6];   // [32]
    const float* W3     = (const float*)d_in[7];   // [32,1]
    const float* b3     = (const float*)d_in[8];   // [1]
    float* out = (float*)d_out;                    // [n,1]

    const int n = in_sizes[0] / 3;
    const int e = in_sizes[2];
    const int* row = ei;
    const int* col = ei + e;

    // workspace layout (floats), ~33 MB total
    float* ws    = (float*)d_ws;
    float* dinv  = ws;                     // [n]     deg -> dinv (in place)
    float* xw    = dinv  + n;              // [n*32]
    float* agg1  = xw    + (size_t)n * 32; // [n*32]
    float* nrm   = agg1  + (size_t)n * 32; // [e]
    float* hw    = nrm   + e;              // [n]
    float* stats = hw    + n;              // [64] sum/sumsq
    float* ss    = stats + 64;             // [64] scale/shift

    const int B = 256;
    const int ntiles = (n + 15) / 16;
    const int wmma_blocks = (ntiles * 32 + B - 1) / B;

    k_deg_init  <<<(n + B - 1) / B, B, 0, stream>>>(dinv, n);
    k_deg_accum <<<(e + B - 1) / B, B, 0, stream>>>(col, ew, dinv, e);
    k_dinv      <<<(n + B - 1) / B, B, 0, stream>>>(dinv, n);
    k_norm      <<<(e + B - 1) / B, B, 0, stream>>>(row, col, ew, dinv, nrm, e);

    k_xw_wmma   <<<wmma_blocks, B, 0, stream>>>(x, W1, xw, n, ntiles);
    k_agg1_init <<<((size_t)n * 32 + B - 1) / B, B, 0, stream>>>(xw, dinv, agg1, n);
    k_scatter1  <<<((size_t)e * 32 + B - 1) / B, B, 0, stream>>>(row, col, nrm, xw, agg1, e);

    k_stats_zero  <<<1, 64, 0, stream>>>(stats);
    k_bn_reduce   <<<512, B, 0, stream>>>(agg1, stats, n);
    k_bn_finalize <<<1, 32, 0, stream>>>(stats, gamma1, beta1, ss, n);

    k_hw_wmma   <<<wmma_blocks, B, 0, stream>>>(agg1, ss, W3, hw, n, ntiles);
    k_out_init  <<<(n + B - 1) / B, B, 0, stream>>>(hw, dinv, b3, out, n);
    k_scatter2  <<<(e + B - 1) / B, B, 0, stream>>>(row, col, nrm, hw, out, e);
}
